// TaskSpecificGIMANSystem_10514079941123
// MI455X (gfx1250) — matrix operand order repaired and checked by reading, not compiled, not run on hardware
//
#include <hip/hip_runtime.h>

// ---------------------------------------------------------------------------
// TaskSpecificGIMANSystem forward for MI455X (gfx1250, wave32, WMMA).
// Flash-style GAT + MHA attention using v_wmma_f32_16x16x32_bf16.
// Operands laid out K-contiguous so h and V WMMA B-fragments are single 32B loads.
// ---------------------------------------------------------------------------

#define Bv 4
#define Nv 2048
#define SDv 256
#define GDv 8
#define TDv 64
#define Ev 32
#define Hv 2
#define Fv 32
#define Dv 96
#define HDv 48

typedef __attribute__((ext_vector_type(16))) __bf16 v16bf;
typedef __attribute__((ext_vector_type(8)))  float  v8f;
typedef __attribute__((ext_vector_type(8)))  int    v8i;

__device__ __forceinline__ unsigned short f2bf(float f) {
  union { float f; unsigned u; } v; v.f = f;
  unsigned r = v.u + 0x7FFFu + ((v.u >> 16) & 1u);   // RNE
  return (unsigned short)(r >> 16);
}
__device__ __forceinline__ __bf16 bfb(unsigned short s) {
  union { unsigned short s; __bf16 b; } v; v.s = s; return v.b;
}
__device__ __forceinline__ v16bf bfzero16() {
  union { v8i i; v16bf v; } u; u.i = (v8i){0,0,0,0,0,0,0,0}; return u.v;
}
// Build A-fragment from two contiguous 16B runs (A layout: slot s<8 -> K=khalf*8+s,
// slot s>=8 -> K=16+khalf*8+(s-8); each run is 8 consecutive ushorts).
__device__ __forceinline__ v16bf ldA(const unsigned short* base, int khalf) {
  union { uint4 q[2]; v16bf v; } u;
  u.q[0] = *(const uint4*)(base + khalf * 8);
  u.q[1] = *(const uint4*)(base + 16 + khalf * 8);
  return u.v;
}
__device__ __forceinline__ float wsum(float v) {
#pragma unroll
  for (int o = 16; o > 0; o >>= 1) v += __shfl_xor(v, o, 32);
  return v;
}
// 16-bit A-matrix K index for slot (r,c), lane-half khalf (contiguous: K = 16*khalf + 2r + c)
__device__ __forceinline__ int kofA(int r, int c, int khalf) {
  int base = 2 * r + c;
  return base + khalf * 16;
}

// ---------------------------------------------------------------------------
// 1) Embeddings: Linear -> LayerNorm(32) -> ReLU, concat into x[B,N,96].
//    One wave per embedding; LN group == wave32.
// ---------------------------------------------------------------------------
__global__ void embed_kernel(const float* __restrict__ sp, const float* __restrict__ ge,
                             const float* __restrict__ te,
                             const float* spW, const float* spB, const float* spG, const float* spBe,
                             const float* geW, const float* geB, const float* geG, const float* geBe,
                             const float* teW, const float* teB, const float* teG, const float* teBe,
                             float* __restrict__ x) {
  int row  = blockIdx.x;            // b*N + n
  int wid  = threadIdx.x >> 5;
  int lane = threadIdx.x & 31;
  float y;
  if (wid == 0) {
    const float* in = sp + (size_t)row * SDv;
    const float* w  = spW + lane * SDv;
    float a = spB[lane];
    for (int k = 0; k < SDv; ++k) a += in[k] * w[k];
    y = a;
  } else if (wid == 1) {
    const float* in = ge + (size_t)row * GDv;
    const float* w  = geW + lane * GDv;
    float a = geB[lane];
    for (int k = 0; k < GDv; ++k) a += in[k] * w[k];
    y = a;
  } else {
    const float* in = te + (size_t)row * TDv;
    const float* w  = teW + lane * TDv;
    float a = teB[lane];
    for (int k = 0; k < TDv; ++k) a += in[k] * w[k];
    y = a;
  }
  float mean = wsum(y) * (1.f / 32.f);
  float d    = y - mean;
  float var  = wsum(d * d) * (1.f / 32.f);
  const float* g  = (wid == 0) ? spG  : (wid == 1) ? geG  : teG;
  const float* be = (wid == 0) ? spBe : (wid == 1) ? geBe : teBe;
  float v = d * rsqrtf(var + 1e-5f) * g[lane] + be[lane];
  x[(size_t)row * Dv + wid * 32 + lane] = fmaxf(v, 0.f);
}

// ---------------------------------------------------------------------------
// 2) GAT features: h = x @ gat_W^T, stored TRANSPOSED bf16  ht[64][B*N]
//    (K-contiguous for WMMA B-fragments); src/dst via wave reduce.
// ---------------------------------------------------------------------------
__global__ void gat_h_kernel(const float* __restrict__ x, const float* __restrict__ gatW,
                             const float* __restrict__ a1, const float* __restrict__ a2,
                             unsigned short* __restrict__ ht,
                             float* __restrict__ srcv, float* __restrict__ dstv) {
  int row  = blockIdx.x;
  int head = threadIdx.x >> 5;
  int lane = threadIdx.x & 31;
  int oc   = head * Fv + lane;
  const float* xr = x + (size_t)row * Dv;
  const float* w  = gatW + oc * Dv;
  float h = 0.f;
  for (int k = 0; k < Dv; ++k) h += xr[k] * w[k];
  ht[(size_t)oc * (Bv * Nv) + row] = f2bf(h);
  float s = wsum(h * a1[oc]);
  float d = wsum(h * a2[oc]);
  if (lane == 0) {
    srcv[row * Hv + head] = s;
    dstv[row * Hv + head] = d;
  }
}

// ---------------------------------------------------------------------------
// 3) GAT attention, flash-style: one wave per (b, head, 16-row tile).
//    Scores built directly in the WMMA A layout; alpha@h via WMMA; h columns
//    loaded as single 32B vectors from the transposed ht.
// ---------------------------------------------------------------------------
__global__ void gat_attn_kernel(const float* __restrict__ srcv, const float* __restrict__ dstv,
                                const unsigned short* __restrict__ ht,
                                const float* __restrict__ adj,
                                float* __restrict__ gat_raw) {
  int wid  = threadIdx.x >> 5;
  int lane = threadIdx.x & 31;
  int t    = blockIdx.x * 4 + wid;      // 0 .. B*H*(N/16)-1
  int b    = t >> 8;
  int head = (t >> 7) & 1;
  int i0   = (t & 127) << 4;
  int lrow  = lane & 15;
  int khalf = lane >> 4;
  int n     = lrow;
  int bN    = b * Nv;
  float src_i = srcv[(bN + i0 + lrow) * Hv + head];
  const float* adjr = adj + (size_t)(i0 + lrow) * Nv;
  const unsigned short* hcol0 = ht + (size_t)(head * Fv + n)      * (Bv * Nv) + bN;
  const unsigned short* hcol1 = ht + (size_t)(head * Fv + 16 + n) * (Bv * Nv) + bN;

  float m = -3.0e38f, l = 0.f;          // l: per-lane partial (reduced at end)
  v8f acc0 = {}, acc1 = {};
  for (int j0 = 0; j0 < Nv; j0 += 32) {
    if (j0 + 32 < Nv) __builtin_prefetch(adjr + j0 + 32, 0, 1);
    float ev[16];
    float mloc = -3.0e38f;
#pragma unroll
    for (int r = 0; r < 8; ++r)
#pragma unroll
      for (int c = 0; c < 2; ++c) {
        int kk = kofA(r, c, khalf);
        int j  = j0 + kk;
        float e = src_i + dstv[(bN + j) * Hv + head];
        e = (e > 0.f) ? e : 0.2f * e;                 // leaky_relu(0.2)
        if (adjr[j] == 0.f) e = -3.0e38f;             // mask
        ev[2 * r + c] = e;
        mloc = fmaxf(mloc, e);
      }
    mloc = fmaxf(mloc, __shfl_xor(mloc, 16, 32));     // pair covers all 32 K
    float mnew  = fmaxf(m, mloc);
    float scale = __expf(m - mnew);
#pragma unroll
    for (int r = 0; r < 8; ++r) { acc0[r] *= scale; acc1[r] *= scale; }
    float psum = 0.f;
    v16bf pa;
#pragma unroll
    for (int q = 0; q < 16; ++q) {
      float p = __expf(ev[q] - mnew);
      psum += p;
      pa[q] = bfb(f2bf(p));
    }
    l = l * scale + psum;               // deferred cross-lane reduction
    m = mnew;
    v16bf hb0 = *(const v16bf*)(hcol0 + j0 + khalf * 16);
    v16bf hb1 = *(const v16bf*)(hcol1 + j0 + khalf * 16);
    acc0 = __builtin_amdgcn_wmma_f32_16x16x32_bf16(false, pa, false, hb0, (short)0, acc0, false, false);
    acc1 = __builtin_amdgcn_wmma_f32_16x16x32_bf16(false, pa, false, hb1, (short)0, acc1, false, false);
  }
  float ltot = l + __shfl_xor(l, 16, 32);             // full row sum (row = lrow)
#pragma unroll
  for (int r = 0; r < 8; ++r) {
    int row  = r + 8 * khalf;                          // C layout: M = r + 8*half
    float inv = __builtin_amdgcn_rcpf(__shfl(ltot, row, 32));
    float* op = gat_raw + (size_t)(bN + i0 + row) * (Hv * Fv) + head * Fv;
    op[n]      = acc0[r] * inv;
    op[16 + n] = acc1[r] * inv;
  }
}

// ---------------------------------------------------------------------------
// 4) gat_out = LN(gat_raw @ oW^T + ob); qkv = gat_out @ iW^T + ib.
//    q,k bf16 [B,H,N,48] (K-contiguous rows); v bf16 TRANSPOSED [B,H,48,N].
//    1/sqrt(48) folded into q.
// ---------------------------------------------------------------------------
__global__ void gat_proj_qkv_kernel(const float* __restrict__ gat_raw,
                                    const float* oW, const float* obp,
                                    const float* gg, const float* gbe,
                                    const float* iW, const float* ib,
                                    unsigned short* __restrict__ qb,
                                    unsigned short* __restrict__ kb,
                                    unsigned short* __restrict__ vt) {
  __shared__ float sg[Dv];
  __shared__ float red[6];
  int row = blockIdx.x;
  int j   = threadIdx.x;
  int wid = j >> 5, lane = j & 31;
  const float* gr = gat_raw + (size_t)row * (Hv * Fv);
  float y = obp[j];
  const float* w = oW + j * (Hv * Fv);
  for (int k = 0; k < Hv * Fv; ++k) y += gr[k] * w[k];
  float ps = wsum(y), ps2 = wsum(y * y);
  if (lane == 0) { red[wid] = ps; red[3 + wid] = ps2; }
  __syncthreads();
  float mean = (red[0] + red[1] + red[2]) * (1.f / Dv);
  float var  = (red[3] + red[4] + red[5]) * (1.f / Dv) - mean * mean;
  sg[j] = (y - mean) * rsqrtf(var + 1e-5f) * gg[j] + gbe[j];
  __syncthreads();
  int b = row / Nv, nn = row % Nv;
  int head = j / HDv, d0 = j % HDv;
  size_t qkidx = (size_t)((b * Hv + head) * Nv + nn) * HDv + d0;
  size_t vidx  = (size_t)((b * Hv + head) * HDv + d0) * Nv + nn;
  const float qscale = 0.14433756729740643f;   // 1/sqrt(48)
#pragma unroll
  for (int tq = 0; tq < 3; ++tq) {
    int jj = tq * Dv + j;
    float v = ib[jj];
    const float* wi = iW + (size_t)jj * Dv;
    for (int k = 0; k < Dv; ++k) v += sg[k] * wi[k];
    if      (tq == 0) qb[qkidx] = f2bf(v * qscale);
    else if (tq == 1) kb[qkidx] = f2bf(v);
    else              vt[vidx]  = f2bf(v);
  }
}

// ---------------------------------------------------------------------------
// 5) MHA flash attention: QK^T (K=48 padded to 64) + PV, all WMMA.
//    Tile-wide running max (scalar stats), per-lane partial row sums reduced
//    once at the end; P converted C->A layout via LDS + s_wait_dscnt.
// ---------------------------------------------------------------------------
__global__ void mha_kernel(const unsigned short* __restrict__ qb,
                           const unsigned short* __restrict__ kb,
                           const unsigned short* __restrict__ vt,
                           float* __restrict__ obuf) {
  __shared__ __align__(32) unsigned short psh[4 * 16 * 32];
  int wid  = threadIdx.x >> 5;
  int lane = threadIdx.x & 31;
  int t    = blockIdx.x * 4 + wid;
  int b    = t >> 8;
  int head = (t >> 7) & 1;
  int i0   = (t & 127) << 4;
  int lrow  = lane & 15;
  int khalf = lane >> 4;
  int n     = lrow;
  size_t hbase = (size_t)(b * Hv + head) * Nv;
  unsigned short* pt = psh + wid * 16 * 32;

  // Q A-fragments (fixed over j loop); K=48 split into 32 + 16(+zero pad)
  const unsigned short* qr = qb + (hbase + i0 + lrow) * HDv;
  v16bf aq0 = ldA(qr, khalf);
  v16bf aq1;
  {
    union { uint4 q[2]; v16bf v; } u;
    u.q[0] = *(const uint4*)(qr + 32 + khalf * 8);  // K=32..47 (slots 0..7)
    u.q[1] = uint4{0, 0, 0, 0};                     // K=48..63 pad
    aq1 = u.v;
  }
  const unsigned short* vcol[3];
#pragma unroll
  for (int g = 0; g < 3; ++g)
    vcol[g] = vt + (size_t)((b * Hv + head) * HDv + g * 16 + n) * Nv;

  float m = -3.0e38f;
  float l8[8];
  v8f accv[3] = {};
#pragma unroll
  for (int r = 0; r < 8; ++r) l8[r] = 0.f;

  for (int j0 = 0; j0 < Nv; j0 += 32) {
    v8f sc[2];
#pragma unroll
    for (int jt = 0; jt < 2; ++jt) {
      const unsigned short* kr = kb + (hbase + j0 + jt * 16 + n) * HDv;
      union { uint4 q[2]; v16bf v; } ub0, ub1;        // K=0..31 in A (table) order
      ub0.q[0] = *(const uint4*)(kr + khalf * 8);
      ub0.q[1] = *(const uint4*)(kr + 16 + khalf * 8);
      ub1.q[0] = *(const uint4*)(kr + 32 + khalf * 8);
      ub1.q[1] = uint4{0, 0, 0, 0};
      v16bf bk0 = ub0.v;
      v16bf bk1 = ub1.v;
      v8f s = {};
      s = __builtin_amdgcn_wmma_f32_16x16x32_bf16(false, aq0, false, bk0, (short)0, s, false, false);
      s = __builtin_amdgcn_wmma_f32_16x16x32_bf16(false, aq1, false, bk1, (short)0, s, false, false);
      sc[jt] = s;
    }
    // tile-wide running max (exact softmax; scalar stats shared by all lanes)
    float mloc = -3.0e38f;
#pragma unroll
    for (int jt = 0; jt < 2; ++jt)
#pragma unroll
      for (int r = 0; r < 8; ++r) mloc = fmaxf(mloc, sc[jt][r]);
#pragma unroll
    for (int off = 1; off < 32; off <<= 1) mloc = fmaxf(mloc, __shfl_xor(mloc, off, 32));
    float mn    = fmaxf(m, mloc);
    float scale = __expf(m - mn);
    m = mn;
#pragma unroll
    for (int g = 0; g < 3; ++g)
#pragma unroll
      for (int r = 0; r < 8; ++r) accv[g][r] *= scale;
#pragma unroll
    for (int r = 0; r < 8; ++r) l8[r] *= scale;
    // P = exp(S - m): per-lane partial row sums; stage tile through LDS
#pragma unroll
    for (int jt = 0; jt < 2; ++jt)
#pragma unroll
      for (int r = 0; r < 8; ++r) {
        int row = r + 8 * khalf;
        float p = __expf(sc[jt][r] - m);
        l8[r] += p;
        pt[row * 32 + jt * 16 + n] = f2bf(p);
      }
    asm volatile("s_wait_dscnt 0" ::: "memory");        // CDNA5 split DS counter
    union { uint4 q[2]; v16bf v; } pu;
    const unsigned short* pr = pt + lrow * 32;
    pu.q[0] = *(const uint4*)(pr + khalf * 16);
    pu.q[1] = *(const uint4*)(pr + khalf * 16 + 8);
    v16bf pa = pu.v;
#pragma unroll
    for (int g = 0; g < 3; ++g) {
      v16bf bv = *(const v16bf*)(vcol[g] + j0 + khalf * 16);
      accv[g] = __builtin_amdgcn_wmma_f32_16x16x32_bf16(false, pa, false, bv, (short)0, accv[g], false, false);
    }
  }
#pragma unroll
  for (int r = 0; r < 8; ++r) {
    float lt = l8[r];
#pragma unroll
    for (int off = 1; off < 16; off <<= 1) lt += __shfl_xor(lt, off, 32);
    float inv = __builtin_amdgcn_rcpf(lt);
    int row = r + 8 * khalf;
#pragma unroll
    for (int g = 0; g < 3; ++g)
      obuf[(size_t)(b * Nv + i0 + row) * Dv + head * HDv + g * 16 + n] = accv[g][r] * inv;
  }
}

// ---------------------------------------------------------------------------
// 6) rep_pre[b, j] = sum_n o[b, n, j]
// ---------------------------------------------------------------------------
__global__ void rowsum_kernel(const float* __restrict__ obuf, float* __restrict__ rep) {
  int b = blockIdx.x, j = threadIdx.x;
  float s = 0.f;
  for (int nn = 0; nn < Nv; ++nn) s += obuf[(size_t)(b * Nv + nn) * Dv + j];
  rep[b * Dv + j] = s;
}

// ---------------------------------------------------------------------------
// 7) Output projection (folded pooling) + towers.
// ---------------------------------------------------------------------------
__device__ float tower96(const float* rv, int j,
                         const float* W1, const float* b1, const float* g1, const float* be1,
                         const float* W2, const float* b2, const float* g2, const float* be2,
                         const float* W3, const float* b3,
                         float* s1, float* s2, float* stats) {
  if (j < 64) {
    float z = b1[j];
    const float* w = W1 + j * Dv;
    for (int k = 0; k < Dv; ++k) z += rv[k] * w[k];
    s1[j] = z;
  }
  __syncthreads();
  if (j == 0) {
    float su = 0.f, sq = 0.f;
    for (int k = 0; k < 64; ++k) { su += s1[k]; sq += s1[k] * s1[k]; }
    float mn = su / 64.f;
    stats[0] = mn; stats[1] = sq / 64.f - mn * mn;
  }
  __syncthreads();
  if (j < 64) s1[j] = fmaxf((s1[j] - stats[0]) * rsqrtf(stats[1] + 1e-5f) * g1[j] + be1[j], 0.f);
  __syncthreads();
  if (j < 32) {
    float z = b2[j];
    const float* w = W2 + j * 64;
    for (int k = 0; k < 64; ++k) z += s1[k] * w[k];
    s2[j] = z;
  }
  __syncthreads();
  if (j == 0) {
    float su = 0.f, sq = 0.f;
    for (int k = 0; k < 32; ++k) { su += s2[k]; sq += s2[k] * s2[k]; }
    float mn = su / 32.f;
    stats[0] = mn; stats[1] = sq / 32.f - mn * mn;
  }
  __syncthreads();
  if (j < 32) s2[j] = fmaxf((s2[j] - stats[0]) * rsqrtf(stats[1] + 1e-5f) * g2[j] + be2[j], 0.f);
  __syncthreads();
  if (j == 0) {
    float r = b3[0];
    for (int k = 0; k < 32; ++k) r += s2[k] * W3[k];
    stats[0] = r;
  }
  __syncthreads();
  return stats[0];
}

__global__ void final_kernel(const float* __restrict__ rep,
                             const float* oW, const float* obp,
                             const float* mtW1, const float* mtb1, const float* mtg1, const float* mtbe1,
                             const float* mtW2, const float* mtb2, const float* mtg2, const float* mtbe2,
                             const float* mtW3, const float* mtb3,
                             const float* ctW1, const float* ctb1, const float* ctg1, const float* ctbe1,
                             const float* ctW2, const float* ctb2, const float* ctg2, const float* ctbe2,
                             const float* ctW3, const float* ctb3,
                             float* __restrict__ out) {
  __shared__ float rv[Dv], s1[64], s2[32], stats[2];
  int j = threadIdx.x;
  for (int b = 0; b < Bv; ++b) {
    // rep = (sum_n o_n) @ oW^T + N*ob
    float v = obp[j] * (float)Nv;
    const float* w  = oW + j * Dv;
    const float* rp = rep + b * Dv;
    for (int k = 0; k < Dv; ++k) v += rp[k] * w[k];
    rv[j] = v;
    __syncthreads();
    float motor = tower96(rv, j, mtW1, mtb1, mtg1, mtbe1, mtW2, mtb2, mtg2, mtbe2, mtW3, mtb3, s1, s2, stats);
    float cg    = tower96(rv, j, ctW1, ctb1, ctg1, ctbe1, ctW2, ctb2, ctg2, ctbe2, ctW3, ctb3, s1, s2, stats);
    if (j == 0) { out[b] = motor; out[Bv + b] = 1.f / (1.f + __expf(-cg)); }
    __syncthreads();
  }
}

// ---------------------------------------------------------------------------
extern "C" void kernel_launch(void* const* d_in, const int* in_sizes, int n_in,
                              void* d_out, int out_size, void* d_ws, size_t ws_size,
                              hipStream_t stream) {
  (void)in_sizes; (void)n_in; (void)out_size; (void)ws_size;
  const float* spatial  = (const float*)d_in[0];
  const float* genomic  = (const float*)d_in[1];
  const float* temporal = (const float*)d_in[2];
  const float* adj      = (const float*)d_in[3];
  // params (dict insertion order)
  const float* sp_W  = (const float*)d_in[4];
  const float* sp_b  = (const float*)d_in[5];
  const float* sp_g  = (const float*)d_in[6];
  const float* sp_be = (const float*)d_in[7];
  const float* ge_W  = (const float*)d_in[8];
  const float* ge_b  = (const float*)d_in[9];
  const float* ge_g  = (const float*)d_in[10];
  const float* ge_be = (const float*)d_in[11];
  const float* te_W  = (const float*)d_in[12];
  const float* te_b  = (const float*)d_in[13];
  const float* te_g  = (const float*)d_in[14];
  const float* te_be = (const float*)d_in[15];
  const float* gat_W  = (const float*)d_in[16];
  const float* gat_a1 = (const float*)d_in[17];
  const float* gat_a2 = (const float*)d_in[18];
  const float* gat_oW = (const float*)d_in[19];
  const float* gat_ob = (const float*)d_in[20];
  const float* gat_g  = (const float*)d_in[21];
  const float* gat_be = (const float*)d_in[22];
  const float* mha_iW = (const float*)d_in[23];
  const float* mha_ib = (const float*)d_in[24];
  const float* mha_oW = (const float*)d_in[25];
  const float* mha_ob = (const float*)d_in[26];
  const float* mt_W1 = (const float*)d_in[27];
  const float* mt_b1 = (const float*)d_in[28];
  const float* mt_g1 = (const float*)d_in[29];
  const float* mt_be1 = (const float*)d_in[30];
  const float* mt_W2 = (const float*)d_in[31];
  const float* mt_b2 = (const float*)d_in[32];
  const float* mt_g2 = (const float*)d_in[33];
  const float* mt_be2 = (const float*)d_in[34];
  const float* mt_W3 = (const float*)d_in[35];
  const float* mt_b3 = (const float*)d_in[36];
  const float* ct_W1 = (const float*)d_in[37];
  const float* ct_b1 = (const float*)d_in[38];
  const float* ct_g1 = (const float*)d_in[39];
  const float* ct_be1 = (const float*)d_in[40];
  const float* ct_W2 = (const float*)d_in[41];
  const float* ct_b2 = (const float*)d_in[42];
  const float* ct_g2 = (const float*)d_in[43];
  const float* ct_be2 = (const float*)d_in[44];
  const float* ct_W3 = (const float*)d_in[45];
  const float* ct_b3 = (const float*)d_in[46];

  // workspace carve (~15 MB)
  char* wp = (char*)d_ws;
  auto carve = [&](size_t bytes) -> char* {
    char* p = wp;
    wp += (bytes + 255) & ~(size_t)255;
    return p;
  };
  float*          x       = (float*)carve((size_t)Bv * Nv * Dv * 4);
  unsigned short* ht      = (unsigned short*)carve((size_t)Hv * Fv * Bv * Nv * 2);
  float*          srcv    = (float*)carve((size_t)Bv * Nv * Hv * 4);
  float*          dstv    = (float*)carve((size_t)Bv * Nv * Hv * 4);
  float*          gat_raw = (float*)carve((size_t)Bv * Nv * Hv * Fv * 4);
  unsigned short* qb      = (unsigned short*)carve((size_t)Bv * Hv * Nv * HDv * 2);
  unsigned short* kb      = (unsigned short*)carve((size_t)Bv * Hv * Nv * HDv * 2);
  unsigned short* vt      = (unsigned short*)carve((size_t)Bv * Hv * HDv * Nv * 2);
  float*          obuf    = (float*)carve((size_t)Bv * Nv * Dv * 4);
  float*          rep     = (float*)carve((size_t)Bv * Dv * 4);

  embed_kernel<<<Bv * Nv, 96, 0, stream>>>(spatial, genomic, temporal,
      sp_W, sp_b, sp_g, sp_be, ge_W, ge_b, ge_g, ge_be, te_W, te_b, te_g, te_be, x);
  gat_h_kernel<<<Bv * Nv, 64, 0, stream>>>(x, gat_W, gat_a1, gat_a2, ht, srcv, dstv);
  gat_attn_kernel<<<(Bv * Hv * (Nv / 16)) / 4, 128, 0, stream>>>(srcv, dstv, ht, adj, gat_raw);
  gat_proj_qkv_kernel<<<Bv * Nv, 96, 0, stream>>>(gat_raw, gat_oW, gat_ob, gat_g, gat_be,
                                                  mha_iW, mha_ib, qb, kb, vt);
  mha_kernel<<<(Bv * Hv * (Nv / 16)) / 4, 128, 0, stream>>>(qb, kb, vt, obuf);
  rowsum_kernel<<<Bv, 96, 0, stream>>>(obuf, rep);
  final_kernel<<<1, 96, 0, stream>>>(rep, mha_oW, mha_ob,
      mt_W1, mt_b1, mt_g1, mt_be1, mt_W2, mt_b2, mt_g2, mt_be2, mt_W3, mt_b3,
      ct_W1, ct_b1, ct_g1, ct_be1, ct_W2, ct_b2, ct_g2, ct_be2, ct_W3, ct_b3,
      (float*)d_out);
}
